// MoE_67018669686847
// MI455X (gfx1250) — matrix operand order repaired and checked by reading, compiled, not run on hardware
//
#include <hip/hip_runtime.h>
#include <hip/hip_bf16.h>

// ---------------------------------------------------------------------------
// MoE forward for MI455X (gfx1250, wave32, WMMA), with real top-2 dispatch:
//   1) router hidden GEMM (bf16 WMMA, fused tanh+bias)
//   2) gating (softmax/top-2) + atomic build of per-expert token lists
//   3) per-expert gathered GEMM1 (fused gelu) and scattered, gate-weighted
//      GEMM2 accumulate -> only ~292 GFLOP instead of the dense 1.1 TFLOP.
// fp32 operands are converted to bf16 while staging into LDS; compute uses
// v_wmma_f32_16x16x32_bf16 with fp32 accumulation. bf16 A tiles (H) use the
// CDNA5 async global->LDS path (ASYNCcnt + s_wait_asynccnt).
// ---------------------------------------------------------------------------

typedef unsigned short u16;
typedef __bf16 bf16x16 __attribute__((ext_vector_type(16)));
typedef __bf16 bf16x8  __attribute__((ext_vector_type(8)));
typedef float  f32x8   __attribute__((ext_vector_type(8)));
typedef int    v4i     __attribute__((ext_vector_type(4)));

#define TILE_M 64
#define TILE_N 128
#define TILE_K 64
#define LDS_PITCH 80   // bf16 elems/row = 160 B (multiple of 32 -> aligned frag loads)

#if __has_builtin(__builtin_amdgcn_global_load_async_to_lds_b128) && \
    __has_builtin(__builtin_amdgcn_s_wait_asynccnt)
#define HAVE_ASYNC_LDS 1
#else
#define HAVE_ASYNC_LDS 0
#endif

__device__ __forceinline__ u16 f2bf(float f) {
  unsigned u = __builtin_bit_cast(unsigned, f);
  u += 0x7FFFu + ((u >> 16) & 1u);   // round-to-nearest-even
  return (u16)(u >> 16);
}
__device__ __forceinline__ unsigned pk_bf16(float a, float b) {
  return (unsigned)f2bf(a) | ((unsigned)f2bf(b) << 16);
}
__device__ __forceinline__ float bf2f(u16 b) {
  unsigned u = ((unsigned)b) << 16;
  return __builtin_bit_cast(float, u);
}
__device__ __forceinline__ float gelu_tanh(float x) {
  float x3 = x * x * x;
  return 0.5f * x * (1.0f + tanhf(0.7978845608028654f * (x + 0.044715f * x3)));
}

enum { EPI_TANH_BIAS = 0, EPI_GELU_GATHER = 1, EPI_MOE_SCATTER = 2 };

// C = epilogue(A[M,K] @ B[K,N]); B row-major fp32 (converted to bf16 in LDS).
// EPI_TANH_BIAS : A fp32 direct rows;  out bf16 = tanh(acc + bias[c])
// EPI_GELU_GATHER: A fp32 rows gathered via idxArr (clamped); out bf16 = gelu
//                  written to compacted rows; block exits if row0 >= *mcount
// EPI_MOE_SCATTER: A bf16 compacted rows; epilogue scatters acc*gate into outf
//                  at token rows (skips tok==Tlim sentinel pads)
template <typename TA, int EPI>
__launch_bounds__(256)
__global__ void gemm_bf16_wmma(const TA* __restrict__ A, const float* __restrict__ B,
                               const float* __restrict__ bias,
                               const float* __restrict__ gates,
                               const int* __restrict__ idxArr,
                               const int* __restrict__ mcount,
                               int expert,
                               u16* __restrict__ outb, float* __restrict__ outf,
                               int N, int K, int Tlim)
{
  constexpr bool IS_BF16_A = (sizeof(TA) == 2);

  const int row0 = blockIdx.y * TILE_M;
  const int col0 = blockIdx.x * TILE_N;

  if constexpr (EPI != EPI_TANH_BIAS) {
    if (row0 >= *mcount) return;     // uniform block-level exit (before any wmma)
  }

  __shared__ __align__(32) u16 As[TILE_M * LDS_PITCH];
  __shared__ __align__(32) u16 Bs[TILE_N * LDS_PITCH];   // transposed: Bs[n][k]
  __shared__ int rowTok[TILE_M];

  const int tid   = threadIdx.x;
  const int lane  = tid & 31;
  const int wave  = tid >> 5;       // 0..7
  const int waveM = wave >> 2;      // 0..1
  const int waveN = wave & 3;       // 0..3
  const int h     = lane >> 4;      // lane half
  const int l16   = lane & 15;

  if constexpr (EPI == EPI_GELU_GATHER) {
    if (tid < TILE_M) {
      int v = idxArr[row0 + tid];
      rowTok[tid] = (v < Tlim) ? v : (Tlim - 1);  // clamp sentinel for safe gather
    }
  } else if constexpr (EPI == EPI_MOE_SCATTER) {
    if (tid < TILE_M) rowTok[tid] = idxArr[row0 + tid];  // raw (sentinel == Tlim)
  }

  auto arow = [&](int r) -> int {
    if constexpr (EPI == EPI_GELU_GATHER) return rowTok[r];
    else return row0 + r;
  };

  f32x8 acc[2][2] = {};

  for (int k0 = 0; k0 < K; k0 += TILE_K) {
    __syncthreads();   // also publishes rowTok on the first iteration

    // ---- stage A tile (TILE_M x TILE_K) ----
    if constexpr (!IS_BF16_A) {
      // fp32 -> bf16: 16B global loads, packed 8B LDS stores
#pragma unroll
      for (int i = 0; i < (TILE_M * TILE_K) / (256 * 4); ++i) {   // 4 iters
        int c4 = tid + i * 256;
        int r  = c4 >> 4;               // TILE_K/4 = 16 chunks per row
        int kc = (c4 & 15) << 2;
        const float4 v = *(const float4*)&A[(size_t)arow(r) * K + (k0 + kc)];
        uint2 pk;
        pk.x = pk_bf16(v.x, v.y);
        pk.y = pk_bf16(v.z, v.w);
        *(uint2*)&As[r * LDS_PITCH + kc] = pk;
      }
    } else {
      // bf16 A: straight 16B copies; async global->LDS (ASYNCcnt-tracked)
#pragma unroll
      for (int i = 0; i < (TILE_M * TILE_K) / (256 * 8); ++i) {   // 2 iters
        int c8 = tid + i * 256;
        int r  = c8 >> 3;               // TILE_K/8 = 8 chunks per row
        int kc = (c8 & 7) << 3;
        const TA* gp = &A[(size_t)(row0 + r) * K + (k0 + kc)];
        u16* lp = &As[r * LDS_PITCH + kc];
#if HAVE_ASYNC_LDS
        v4i* gpv = (v4i*)const_cast<TA*>(gp);
        v4i* lpv = (v4i*)lp;
        __builtin_amdgcn_global_load_async_to_lds_b128(
            (__attribute__((address_space(1))) v4i*)gpv,
            (__attribute__((address_space(3))) v4i*)lpv, 0, 0);
#else
        *(uint4*)lp = *(const uint4*)gp;
#endif
      }
    }

    // ---- stage B tile (TILE_K x TILE_N) transposed into Bs[n][k] ----
#pragma unroll
    for (int i = 0; i < (TILE_K * TILE_N) / (256 * 4); ++i) {     // 8 iters
      int c4 = tid + i * 256;
      int k  = c4 >> 5;                 // TILE_N/4 = 32 chunks per k-row
      int nc = (c4 & 31) << 2;
      const float4 v = *(const float4*)&B[(size_t)(k0 + k) * N + (col0 + nc)];
      Bs[(nc + 0) * LDS_PITCH + k] = f2bf(v.x);
      Bs[(nc + 1) * LDS_PITCH + k] = f2bf(v.y);
      Bs[(nc + 2) * LDS_PITCH + k] = f2bf(v.z);
      Bs[(nc + 3) * LDS_PITCH + k] = f2bf(v.w);
    }

    // next K-tile cache prefetch (global_prefetch_b8)
    if (k0 + TILE_K < K) {
      __builtin_prefetch(&B[(size_t)(k0 + TILE_K + (tid >> 2)) * N + col0], 0, 1);
    }

#if HAVE_ASYNC_LDS
    if constexpr (IS_BF16_A) __builtin_amdgcn_s_wait_asynccnt(0);
#endif
    __syncthreads();

    // ---- 2x2 WMMA per wave, two K=32 steps ----
#pragma unroll
    for (int kk = 0; kk < TILE_K; kk += 32) {
      // A frag (16-bit A 16x32): lane half h holds K=[h*16,h*16+16)
      const int ra0 = (waveM * 32 + l16) * LDS_PITCH + kk;
      const int ra1 = (waveM * 32 + 16 + l16) * LDS_PITCH + kk;
      bf16x8 a0lo = *(const bf16x8*)&As[ra0 + h * 16];
      bf16x8 a0hi = *(const bf16x8*)&As[ra0 + 8 + h * 16];
      bf16x8 a1lo = *(const bf16x8*)&As[ra1 + h * 16];
      bf16x8 a1hi = *(const bf16x8*)&As[ra1 + 8 + h * 16];
      bf16x16 afrag[2] = {
        __builtin_shufflevector(a0lo, a0hi, 0,1,2,3,4,5,6,7,8,9,10,11,12,13,14,15),
        __builtin_shufflevector(a1lo, a1hi, 0,1,2,3,4,5,6,7,8,9,10,11,12,13,14,15)
      };
      // B frag (16-bit B 32x16): lane holds column l16, K=[h*16,h*16+16)
      const int cb0 = (waveN * 32 + l16) * LDS_PITCH + kk + h * 16;
      const int cb1 = (waveN * 32 + 16 + l16) * LDS_PITCH + kk + h * 16;
      bf16x16 bfrag[2] = {
        *(const bf16x16*)&Bs[cb0],
        *(const bf16x16*)&Bs[cb1]
      };
#pragma unroll
      for (int ms = 0; ms < 2; ++ms)
#pragma unroll
        for (int ns = 0; ns < 2; ++ns)
          acc[ms][ns] = __builtin_amdgcn_wmma_f32_16x16x32_bf16(
              false, afrag[ms], false, bfrag[ns], (short)0, acc[ms][ns],
              false, false);
    }
  }

  // ---- epilogue. C/D layout: VGPR j -> M = j + 8*(lane>>4); N = lane&15 ----
#pragma unroll
  for (int ms = 0; ms < 2; ++ms) {
#pragma unroll
    for (int ns = 0; ns < 2; ++ns) {
#pragma unroll
      for (int j = 0; j < 8; ++j) {
        int rloc = waveM * 32 + ms * 16 + j + 8 * h;   // row within tile
        int c    = col0 + waveN * 32 + ns * 16 + l16;
        float v  = acc[ms][ns][j];
        if constexpr (EPI == EPI_TANH_BIAS) {
          outb[(size_t)(row0 + rloc) * N + c] = f2bf(tanhf(v + bias[c]));
        } else if constexpr (EPI == EPI_GELU_GATHER) {
          outb[(size_t)(row0 + rloc) * N + c] = f2bf(gelu_tanh(v));
        } else {
          int tok = rowTok[rloc];
          if (tok < Tlim)   // skip sentinel pads; divergence is fine post-WMMA
            outf[(size_t)tok * N + c] += v * gates[tok * 8 + expert];
        }
      }
    }
  }
}

// One thread per token: logits = Hr @ Wg2, softmax over 8, top-2, normalized
// gates; atomically appends the token to its two experts' dispatch lists.
__global__ void router_gate(const u16* __restrict__ Hr, const float* __restrict__ Wg2,
                            float* __restrict__ gates, int* __restrict__ cnt,
                            int* __restrict__ idx, int T, int K, int CAP)
{
  int t = blockIdx.x * blockDim.x + threadIdx.x;
  if (t >= T) return;
  float lg[8] = {0.f, 0.f, 0.f, 0.f, 0.f, 0.f, 0.f, 0.f};
  const u16* hp = Hr + (size_t)t * K;
  for (int k = 0; k < K; ++k) {
    float hv = bf2f(hp[k]);
#pragma unroll
    for (int e = 0; e < 8; ++e) lg[e] = fmaf(hv, Wg2[k * 8 + e], lg[e]);
  }
  float mx = lg[0];
#pragma unroll
  for (int e = 1; e < 8; ++e) mx = fmaxf(mx, lg[e]);
  float p[8], s = 0.f;
#pragma unroll
  for (int e = 0; e < 8; ++e) { p[e] = __expf(lg[e] - mx); s += p[e]; }
  float inv_s = 1.f / s;
#pragma unroll
  for (int e = 0; e < 8; ++e) p[e] *= inv_s;
  int i0 = 0;
#pragma unroll
  for (int e = 1; e < 8; ++e) if (p[e] > p[i0]) i0 = e;
  int i1 = (i0 == 0) ? 1 : 0;
#pragma unroll
  for (int e = 0; e < 8; ++e) if (e != i0 && p[e] > p[i1]) i1 = e;
  float g0 = p[i0], g1 = p[i1];
  float invg = 1.f / (g0 + g1 + 1e-6f);
#pragma unroll
  for (int e = 0; e < 8; ++e) {
    float g = (e == i0) ? g0 * invg : ((e == i1) ? g1 * invg : 0.f);
    gates[t * 8 + e] = g;
  }
  int p0 = atomicAdd(&cnt[i0], 1);
  idx[(size_t)i0 * CAP + p0] = t;
  int p1 = atomicAdd(&cnt[i1], 1);
  idx[(size_t)i1 * CAP + p1] = t;
}

// Pad each expert's token list up to a multiple of TILE_M with sentinel T.
__global__ void pad_expert_counts(const int* __restrict__ cnt, int* __restrict__ cntPad,
                                  int* __restrict__ idx, int T, int CAP)
{
  for (int e = 0; e < 8; ++e) {
    int c  = cnt[e];
    int cp = (c + TILE_M - 1) & ~(TILE_M - 1);
    for (int i = c + (int)threadIdx.x; i < cp; i += (int)blockDim.x)
      idx[(size_t)e * CAP + i] = T;   // sentinel
    if (threadIdx.x == 0) cntPad[e] = cp;
  }
}

__global__ void zero_f32(float* __restrict__ p, int n)
{
  int i = blockIdx.x * blockDim.x + threadIdx.x;
  if (i < n) p[i] = 0.f;
}
__global__ void zero_i32(int* __restrict__ p, int n)
{
  int i = blockIdx.x * blockDim.x + threadIdx.x;
  if (i < n) p[i] = 0;
}

// ---------------------------------------------------------------------------
// Host launcher. Inputs: x, Wg1, bg1, Wg2, W1, W2 (fp32).
// ws: Hr bf16[T,1024] | gates f32[T,8] | cnt | cntPad | idx[8,T] | H bf16[T,4096]
// ---------------------------------------------------------------------------
extern "C" void kernel_launch(void* const* d_in, const int* in_sizes, int n_in,
                              void* d_out, int out_size, void* d_ws, size_t ws_size,
                              hipStream_t stream)
{
  (void)in_sizes; (void)n_in; (void)out_size; (void)ws_size;
  const float* x   = (const float*)d_in[0];
  const float* Wg1 = (const float*)d_in[1];
  const float* bg1 = (const float*)d_in[2];
  const float* Wg2 = (const float*)d_in[3];
  const float* W1  = (const float*)d_in[4];
  const float* W2  = (const float*)d_in[5];
  float* out = (float*)d_out;

  const int T = 4 * 2048;   // 8192 tokens
  const int D = 1024;
  const int F = 4096;
  const int CAP = T;        // per-expert list capacity (count <= T, T % 64 == 0)

  char* ws = (char*)d_ws;
  size_t off = 0;
  u16*   Hr     = (u16*)(ws + off);  off += (size_t)T * D * sizeof(u16);
  float* gates  = (float*)(ws + off); off += (size_t)T * 8 * sizeof(float);
  int*   cnt    = (int*)(ws + off);   off += 256;
  int*   cntPad = (int*)(ws + off);   off += 256;
  int*   idx    = (int*)(ws + off);   off += (size_t)8 * CAP * sizeof(int);
  u16*   H      = (u16*)(ws + off);   // T*F bf16 (compacted per-expert, reused)

  dim3 blk(256);

  // 1) reset dispatch counters
  zero_i32<<<dim3(1), dim3(32), 0, stream>>>(cnt, 8);

  // 2) router hidden: Hr = tanh(x @ Wg1 + bg1), bf16
  gemm_bf16_wmma<float, EPI_TANH_BIAS>
      <<<dim3(D / TILE_N, T / TILE_M), blk, 0, stream>>>(
          x, Wg1, bg1, nullptr, nullptr, nullptr, 0, Hr, nullptr, D, D, T);

  // 3) gating + dispatch-list build
  router_gate<<<dim3(T / 256), blk, 0, stream>>>(Hr, Wg2, gates, cnt, idx, T, D, CAP);

  // 4) pad lists to tile multiples
  pad_expert_counts<<<dim3(1), blk, 0, stream>>>(cnt, cntPad, idx, T, CAP);

  // 5) zero fp32 output accumulator
  zero_f32<<<dim3((T * D + 255) / 256), blk, 0, stream>>>(out, T * D);

  // 6) per-expert gathered FFN + scattered gate-weighted combine
  for (int e = 0; e < 8; ++e) {
    const float* W1e = W1 + (size_t)e * D * F;
    const float* W2e = W2 + (size_t)e * F * D;
    gemm_bf16_wmma<float, EPI_GELU_GATHER>
        <<<dim3(F / TILE_N, T / TILE_M), blk, 0, stream>>>(
            x, W1e, nullptr, nullptr, idx + (size_t)e * CAP, cntPad + e, e,
            H, nullptr, F, D, T);
    gemm_bf16_wmma<u16, EPI_MOE_SCATTER>
        <<<dim3(D / TILE_N, T / TILE_M), blk, 0, stream>>>(
            H, W2e, nullptr, gates, idx + (size_t)e * CAP, cntPad + e, e,
            nullptr, out, D, F, T);
  }
}
